// T5Attention_20658792693936
// MI455X (gfx1250) — matrix operand order repaired and checked by reading, not compiled, not run on hardware
//
#include <hip/hip_runtime.h>
#include <hip/hip_bf16.h>

#define B_   2
#define L_   2048
#define D_   1024
#define H_   16
#define DK_  64
#define XL_  512
#define SEC_ 512
#define KL_  3072   // SEC + XL + L
#define HD_  1024   // H * DK
#define NEGF (-3.0e38f)

typedef __attribute__((ext_vector_type(16))) __bf16 v16bf;
typedef __attribute__((ext_vector_type(8)))  float  v8f;

__device__ inline unsigned short f32_to_bf16(float f) {
  unsigned int u = __builtin_bit_cast(unsigned int, f);
  u += 0x7FFFu + ((u >> 16) & 1u);          // round-to-nearest-even
  return (unsigned short)(u >> 16);
}

union FragU { unsigned int u[8]; v16bf v; };

__device__ inline v8f v8f_zero() {
  v8f z;
#pragma unroll
  for (int i = 0; i < 8; ++i) z[i] = 0.0f;
  return z;
}

__device__ inline v8f wmma_bf16(v16bf a, v16bf b, v8f c) {
  return __builtin_amdgcn_wmma_f32_16x16x32_bf16(false, a, false, b, (short)0, c,
                                                 false, false);
}

// CDNA5 async global->LDS copy (ASYNCcnt-tracked, CDNA5 ISA).
// GVS mode: LDS target taken from the ldsoff VGPR; global source is the SGPR base plus the goff VGPR.
__device__ inline void async_load_b64(unsigned int lds_byte_off,
                                      unsigned int gl_byte_off,
                                      const void* sbase) {
  asm volatile("global_load_async_to_lds_b64 %0, %1, %2"
               :: "v"(lds_byte_off), "v"(gl_byte_off), "s"(sbase)
               : "memory");
}
__device__ inline void wait_async0() {
  asm volatile("s_wait_asynccnt 0" ::: "memory");
}

// LDS byte offset of a generic pointer into __shared__ (flat addr low 32 bits).
__device__ inline unsigned int lds_off(const void* p) {
  return (unsigned int)(size_t)p;
}

// A fragment (16 rows x 32 K, bf16) from [m][k] layout, row stride in ushorts.
// lanes 0-15 -> M, VGPR p sub s -> K = 16*half + 2p + s.
__device__ inline v16bf load_frag_a(const unsigned short* base, int stride,
                                    int m0, int k0, int lane) {
  const int half = lane >> 4;
  const unsigned short* p = base + (size_t)(m0 + (lane & 15)) * stride + k0;
  FragU f;
#pragma unroll
  for (int i = 0; i < 8; ++i)
    f.u[i] = *(const unsigned int*)(p + 16 * half + 2 * i);
  return f.v;
}

// B fragment (32 K x 16 cols, bf16) from [n][k] layout (column-major staged).
// Contiguous k order (same as A): lanes 0-15 hold K=0-15 (k = 2p+s), lanes 16-31 hold K=16-31.
__device__ inline v16bf load_frag_b(const unsigned short* base, int stride,
                                    int n0, int k0, int lane) {
  const int half = lane >> 4;
  const unsigned short* p = base + (size_t)(n0 + (lane & 15)) * stride + k0 + 16 * half;
  FragU f;
#pragma unroll
  for (int i = 0; i < 8; ++i)
    f.u[i] = *(const unsigned int*)(p + 2 * i);
  return f.v;
}

// ---------------------------------------------------------------------------
// bf16 WMMA GEMM core: block tile 128(M) x 256(N), K step 32. 256 threads =
// 8 waves in 2x4, wave tile 64x64 = 16 WMMAs per K step. A is bf16 [M][K]
// row-major, Bt is bf16 [N][K] (pre-transposed). Tiles staged with
// global_load_async_to_lds_b64 (rows are 64B, LDS row stride 72B, 8B aligned).
// ---------------------------------------------------------------------------
__device__ inline void gemm_core_bf16(const unsigned short* __restrict__ A,
                                      const unsigned short* __restrict__ Bt,
                                      int K, int m0, int n0,
                                      unsigned short* lA, unsigned short* lB,
                                      v8f (&acc)[4][4]) {
  const int tid  = threadIdx.x;
  const int lane = tid & 31;
  const int wave = tid >> 5;
  const int wm = (wave >> 2) * 64;
  const int wn = (wave & 3) * 64;
  const unsigned int lAoff = lds_off(lA);
  const unsigned int lBoff = lds_off(lB);

  for (int kb = 0; kb < K; kb += 32) {
    // stage A: 128 rows x 32 bf16 (8 chunks of 8B per row)
#pragma unroll
    for (int t = 0; t < 4; ++t) {
      int c = tid + t * 256;             // 0..1023
      int row = c >> 3, ch = c & 7;
      async_load_b64(lAoff + (unsigned int)(row * 36 + ch * 4) * 2u,
                     (unsigned int)(((m0 + row) * K + kb + ch * 4) * 2),
                     A);
    }
    // stage B: 256 rows x 32 bf16
#pragma unroll
    for (int t = 0; t < 8; ++t) {
      int c = tid + t * 256;             // 0..2047
      int row = c >> 3, ch = c & 7;
      async_load_b64(lBoff + (unsigned int)(row * 36 + ch * 4) * 2u,
                     (unsigned int)(((n0 + row) * K + kb + ch * 4) * 2),
                     Bt);
    }
    wait_async0();
    __syncthreads();

    v16bf af[4];
#pragma unroll
    for (int mi = 0; mi < 4; ++mi)
      af[mi] = load_frag_a(lA, 36, wm + mi * 16, 0, lane);
#pragma unroll
    for (int ni = 0; ni < 4; ++ni) {
      v16bf bfr = load_frag_b(lB, 36, wn + ni * 16, 0, lane);
#pragma unroll
      for (int mi = 0; mi < 4; ++mi)
        acc[mi][ni] = wmma_bf16(af[mi], bfr, acc[mi][ni]);
    }
    __syncthreads();
  }
}

// ---------------------------------------------------------------------------
// Pre-conversion kernels: hidden f32->bf16; weights f32->bf16 transposed [N][K].
// ---------------------------------------------------------------------------
__global__ void __launch_bounds__(256)
convert_hidden_kernel(const float* __restrict__ src, unsigned short* __restrict__ dst) {
  int i = blockIdx.x * 256 + threadIdx.x;
  dst[i] = f32_to_bf16(src[i]);
}

__global__ void __launch_bounds__(256)
convert_wt_kernel(const float* __restrict__ Wq, const float* __restrict__ Wk,
                  const float* __restrict__ Wv, const float* __restrict__ Wo,
                  unsigned short* __restrict__ WqT, unsigned short* __restrict__ WkT,
                  unsigned short* __restrict__ WvT, unsigned short* __restrict__ WoT) {
  int i = blockIdx.x * 256 + threadIdx.x;       // 4 * 1M
  int w = i >> 20;
  int r = i & ((1 << 20) - 1);
  int n = r >> 10, k = r & 1023;
  const float* S = (w == 0) ? Wq : (w == 1) ? Wk : (w == 2) ? Wv : Wo;
  unsigned short* Dp = (w == 0) ? WqT : (w == 1) ? WkT : (w == 2) ? WvT : WoT;
  Dp[r] = f32_to_bf16(S[(size_t)k * HD_ + n]);   // WT[n][k] = W[k][n]
}

// ---------------------------------------------------------------------------
// K1: QKV projections. z=0 -> Q (head-major bf16), z=1/2 -> K/V scattered into
// concatenated [B,H,KL,DK] buffers at offset SEC+XL.
// ---------------------------------------------------------------------------
__global__ void __launch_bounds__(256)
gemm_qkv_kernel(const unsigned short* __restrict__ hidden_bf,
                const unsigned short* __restrict__ WqT,
                const unsigned short* __restrict__ WkT,
                const unsigned short* __restrict__ WvT,
                unsigned short* __restrict__ qout,
                unsigned short* __restrict__ k_all,
                unsigned short* __restrict__ v_all) {
  __shared__ __align__(16) unsigned short lA[128 * 36];
  __shared__ __align__(16) unsigned short lB[256 * 36];
  const int z = blockIdx.z;
  const unsigned short* Wt = (z == 0) ? WqT : (z == 1) ? WkT : WvT;
  const int m0 = blockIdx.y * 128;
  const int n0 = blockIdx.x * 256;
  v8f acc[4][4];
#pragma unroll
  for (int mi = 0; mi < 4; ++mi)
#pragma unroll
    for (int ni = 0; ni < 4; ++ni) acc[mi][ni] = v8f_zero();

  gemm_core_bf16(hidden_bf, Wt, D_, m0, n0, lA, lB, acc);

  const int lane = threadIdx.x & 31;
  const int wave = threadIdx.x >> 5;
  const int wm = (wave >> 2) * 64;
  const int wn = (wave & 3) * 64;
  const int half = lane >> 4;
  const int nlan = lane & 15;
#pragma unroll
  for (int mi = 0; mi < 4; ++mi)
#pragma unroll
    for (int ni = 0; ni < 4; ++ni)
#pragma unroll
      for (int r = 0; r < 8; ++r) {
        int m = m0 + wm + mi * 16 + r + 8 * half;  // b*L + l
        int n = n0 + wn + ni * 16 + nlan;          // h*DK + dk
        int b = m >> 11, l = m & (L_ - 1);
        int h = n >> 6,  dk = n & (DK_ - 1);
        unsigned short v = f32_to_bf16(acc[mi][ni][r]);
        if (z == 0) {
          qout[(((size_t)b * H_ + h) * L_ + l) * DK_ + dk] = v;
        } else {
          unsigned short* dst = (z == 1) ? k_all : v_all;
          dst[(((size_t)b * H_ + h) * KL_ + (SEC_ + XL_) + l) * DK_ + dk] = v;
        }
      }
}

// ---------------------------------------------------------------------------
// K2: convert sec/xl caches f32 -> bf16 into the concatenated K/V buffers.
// ---------------------------------------------------------------------------
__global__ void __launch_bounds__(256)
cache_fill_kernel(const float* __restrict__ xl_k, const float* __restrict__ xl_v,
                  const float* __restrict__ sec_k, const float* __restrict__ sec_v,
                  unsigned short* __restrict__ k_all,
                  unsigned short* __restrict__ v_all) {
  const int per = B_ * H_ * (SEC_ + XL_) * DK_;   // 2097152
  int i = blockIdx.x * 256 + threadIdx.x;
  bool isv = i >= per;
  int j = isv ? i - per : i;
  int dk  = j & 63;
  int pos = (j >> 6) & 1023;
  int bh  = j >> 16;
  float val;
  if (pos < SEC_) {
    const float* s = isv ? sec_v : sec_k;
    val = s[((size_t)bh * SEC_ + pos) * DK_ + dk];
  } else {
    const float* s = isv ? xl_v : xl_k;
    val = s[((size_t)bh * XL_ + (pos - SEC_)) * DK_ + dk];
  }
  unsigned short* dst = isv ? v_all : k_all;
  dst[((size_t)bh * KL_ + pos) * DK_ + dk] = f32_to_bf16(val);
}

// ---------------------------------------------------------------------------
// K3: flash attention. Block = (b,h), 8 waves x 16 queries. 32-key chunks.
// K tile staged with async global->LDS; V staged manually (transposed).
// ---------------------------------------------------------------------------
__global__ void __launch_bounds__(256)
attn_kernel(const unsigned short* __restrict__ q,
            const unsigned short* __restrict__ k_all,
            const unsigned short* __restrict__ v_all,
            const int* __restrict__ doc_ids,
            const int* __restrict__ xl_doc,
            const int* __restrict__ sec_doc,
            const float* __restrict__ rel_bias,
            const float* __restrict__ sec_bias,
            unsigned short* __restrict__ ctx_bf) {
  __shared__ __align__(16) unsigned short k_lds[32 * 68];      // [key][dk], 136B rows
  __shared__ __align__(16) unsigned short vt_lds[64 * 34];     // [dk][key]
  __shared__ __align__(16) unsigned short p_lds[8 * 16 * 34];  // per-wave P scratch
  __shared__ int   kd_lds[32];
  __shared__ float rb_lds[32];

  const int tid  = threadIdx.x;
  const int lane = tid & 31;
  const int wave = tid >> 5;
  const int half = lane >> 4;
  const int nlan = lane & 15;

  const int b  = blockIdx.z;
  const int h  = blockIdx.y;
  const int bh = b * H_ + h;
  const int lw = blockIdx.x * 128 + wave * 16;   // this wave's query base

  if (tid < 32) rb_lds[tid] = rel_bias[tid * H_ + h];
  const float secb = sec_bias[h];
  const unsigned int kldsoff = lds_off(k_lds);

  // persistent Q fragments (dk 0-31 and 32-63)
  const unsigned short* qbase = q + ((size_t)bh * L_ + lw) * DK_;
  v16bf qa0 = load_frag_a(qbase, DK_, 0, 0, lane);
  v16bf qa1 = load_frag_a(qbase, DK_, 0, 32, lane);

  int docq[8];
#pragma unroll
  for (int r = 0; r < 8; ++r)
    docq[r] = doc_ids[b * L_ + lw + r + 8 * half];

  float mrow[8], lrow[8];
#pragma unroll
  for (int r = 0; r < 8; ++r) { mrow[r] = NEGF; lrow[r] = 0.0f; }
  v8f o0 = v8f_zero(), o1 = v8f_zero(), o2 = v8f_zero(), o3 = v8f_zero();

  unsigned short* pw = p_lds + wave * 16 * 34;

  for (int kc = 0; kc < KL_; kc += 32) {
    // ---- async stage of K tile: 32 rows x 128B (16 x 8B chunks per row)
#pragma unroll
    for (int t = 0; t < 2; ++t) {
      int c = tid + t * 256;                    // 0..511
      int row = c >> 4, ch = c & 15;
      async_load_b64(kldsoff + (unsigned int)(row * 68 + ch * 4) * 2u,
                     (unsigned int)((((bh * KL_ + kc + row) * DK_) + ch * 4) * 2),
                     k_all);
    }
    // ---- manual transposed stage of V tile + key docs
    {
      int key = tid >> 3;
      int dkb = (tid & 7) * 8;
      const unsigned short* srcv = v_all + ((size_t)bh * KL_ + kc + key) * DK_ + dkb;
#pragma unroll
      for (int j = 0; j < 8; ++j)
        vt_lds[(dkb + j) * 34 + key] = srcv[j];
      if (tid < 32) {
        int kk = kc + tid;
        int dv;
        if (kk < SEC_)            dv = sec_doc[b * SEC_ + kk];
        else if (kk < SEC_ + XL_) dv = xl_doc[b * XL_ + (kk - SEC_)];
        else                      dv = doc_ids[b * L_ + (kk - SEC_ - XL_)];
        kd_lds[tid] = dv;
      }
    }
    wait_async0();
    __syncthreads();

    // ---- scores: S = Q * K^T  (contraction over dk=64 -> 2 WMMAs per tile)
    v16bf kb0 = load_frag_b(k_lds, 68, 0, 0, lane);
    v16bf kb1 = load_frag_b(k_lds, 68, 0, 32, lane);
    v8f s0 = wmma_bf16(qa0, kb0, v8f_zero());
    s0     = wmma_bf16(qa1, kb1, s0);
    v16bf kb2 = load_frag_b(k_lds, 68, 16, 0, lane);
    v16bf kb3 = load_frag_b(k_lds, 68, 16, 32, lane);
    v8f s1 = wmma_bf16(qa0, kb2, v8f_zero());
    s1     = wmma_bf16(qa1, kb3, s1);

    // ---- T5 relative bias + sec bias + causal + document masks
#pragma unroll
    for (int t = 0; t < 2; ++t) {
      v8f* sp = t ? &s1 : &s0;
      const int  key   = kc + t * 16 + nlan;
      const int  dock  = kd_lds[t * 16 + nlan];
      const bool issec = key < SEC_;
      const float secadd = issec ? secb : 0.0f;
#pragma unroll
      for (int r = 0; r < 8; ++r) {
        int lq = lw + r + 8 * half;
        int delta = (SEC_ + XL_) + lq - key;   // qpos - kpos
        int bucket;
        if (issec) {
          bucket = 31;
        } else {
          int rp = delta > 0 ? delta : 0;
          if (rp < 16) bucket = rp;
          else {
            // 16 + floor(log(rp/16)/log(8) * 16), clamped to 31
            int bl = 16 + (int)(__logf((float)rp * 0.0625f) * 7.69437176f);
            bucket = bl < 31 ? bl : 31;
          }
        }
        float sv = (*sp)[r] + rb_lds[bucket] + secadd;
        if (delta < 0 || dock != docq[r]) sv = NEGF;
        (*sp)[r] = sv;
      }
    }

    // ---- online softmax (row reductions across 16 lanes of a half-wave)
#pragma unroll
    for (int r = 0; r < 8; ++r) {
      float loc = fmaxf(s0[r], s1[r]);
      loc = fmaxf(loc, __shfl_xor(loc, 1, 32));
      loc = fmaxf(loc, __shfl_xor(loc, 2, 32));
      loc = fmaxf(loc, __shfl_xor(loc, 4, 32));
      loc = fmaxf(loc, __shfl_xor(loc, 8, 32));
      float mnew = fmaxf(mrow[r], loc);
      float sc = __expf(mrow[r] - mnew);
      float p0 = __expf(s0[r] - mnew);
      float p1 = __expf(s1[r] - mnew);
      float rs = p0 + p1;
      rs += __shfl_xor(rs, 1, 32);
      rs += __shfl_xor(rs, 2, 32);
      rs += __shfl_xor(rs, 4, 32);
      rs += __shfl_xor(rs, 8, 32);
      lrow[r] = lrow[r] * sc + rs;
      mrow[r] = mnew;
      o0[r] *= sc; o1[r] *= sc; o2[r] *= sc; o3[r] *= sc;
      // C-layout (n=lane) -> A-layout (m=lane) transpose via LDS scratch
      int prow = (r + 8 * half) * 34;
      pw[prow + nlan]      = f32_to_bf16(p0);
      pw[prow + 16 + nlan] = f32_to_bf16(p1);
    }

    asm volatile("s_wait_dscnt 0" ::: "memory");

    // ---- O += P * V  (contraction over 32 keys, 4 dk-column tiles)
    v16bf pa  = load_frag_a(pw, 34, 0, 0, lane);
    v16bf vb0 = load_frag_b(vt_lds, 34, 0,  0, lane);
    v16bf vb1 = load_frag_b(vt_lds, 34, 16, 0, lane);
    v16bf vb2 = load_frag_b(vt_lds, 34, 32, 0, lane);
    v16bf vb3 = load_frag_b(vt_lds, 34, 48, 0, lane);
    o0 = wmma_bf16(pa, vb0, o0);
    o1 = wmma_bf16(pa, vb1, o1);
    o2 = wmma_bf16(pa, vb2, o2);
    o3 = wmma_bf16(pa, vb3, o3);

    __syncthreads();
  }

  // ---- normalize and write ctx [B*L, H*DK] bf16
#pragma unroll
  for (int r = 0; r < 8; ++r) {
    float inv = 1.0f / lrow[r];
    int lq = lw + r + 8 * half;
    unsigned short* dst = ctx_bf + ((size_t)b * L_ + lq) * HD_ + h * DK_ + nlan;
    dst[0]  = f32_to_bf16(o0[r] * inv);
    dst[16] = f32_to_bf16(o1[r] * inv);
    dst[32] = f32_to_bf16(o2[r] * inv);
    dst[48] = f32_to_bf16(o3[r] * inv);
  }
}

// ---------------------------------------------------------------------------
// K4: output projection ctx(bf16) x WoT(bf16) -> out f32 [B,L,D]
// ---------------------------------------------------------------------------
__global__ void __launch_bounds__(256)
gemm_out_kernel(const unsigned short* __restrict__ ctx_bf,
                const unsigned short* __restrict__ WoT,
                float* __restrict__ out) {
  __shared__ __align__(16) unsigned short lA[128 * 36];
  __shared__ __align__(16) unsigned short lB[256 * 36];
  const int m0 = blockIdx.y * 128;
  const int n0 = blockIdx.x * 256;
  v8f acc[4][4];
#pragma unroll
  for (int mi = 0; mi < 4; ++mi)
#pragma unroll
    for (int ni = 0; ni < 4; ++ni) acc[mi][ni] = v8f_zero();

  gemm_core_bf16(ctx_bf, WoT, HD_, m0, n0, lA, lB, acc);

  const int lane = threadIdx.x & 31;
  const int wave = threadIdx.x >> 5;
  const int wm = (wave >> 2) * 64;
  const int wn = (wave & 3) * 64;
  const int half = lane >> 4;
  const int nlan = lane & 15;
#pragma unroll
  for (int mi = 0; mi < 4; ++mi)
#pragma unroll
    for (int ni = 0; ni < 4; ++ni)
#pragma unroll
      for (int r = 0; r < 8; ++r) {
        int m = m0 + wm + mi * 16 + r + 8 * half;
        int n = n0 + wn + ni * 16 + nlan;
        out[(size_t)m * D_ + n] = acc[mi][ni][r];
      }
}

// ---------------------------------------------------------------------------
extern "C" void kernel_launch(void* const* d_in, const int* in_sizes, int n_in,
                              void* d_out, int out_size, void* d_ws, size_t ws_size,
                              hipStream_t stream) {
  (void)in_sizes; (void)n_in; (void)out_size; (void)ws_size;
  const float* hidden  = (const float*)d_in[0];
  const float* xl_k    = (const float*)d_in[1];
  const float* xl_v    = (const float*)d_in[2];
  const float* sec_k   = (const float*)d_in[3];
  const float* sec_v   = (const float*)d_in[4];
  const int*   doc_ids = (const int*)d_in[5];
  const int*   xl_doc  = (const int*)d_in[6];
  const int*   sec_doc = (const int*)d_in[7];
  const float* Wq      = (const float*)d_in[8];
  const float* Wk      = (const float*)d_in[9];
  const float* Wv      = (const float*)d_in[10];
  const float* Wo      = (const float*)d_in[11];
  const float* relb    = (const float*)d_in[12];
  const float* secbias = (const float*)d_in[13];
  float* out = (float*)d_out;

  // workspace layout (56 MiB total)
  char* ws = (char*)d_ws;
  const size_t MiB = 1024 * 1024;
  unsigned short* hid_bf = (unsigned short*)(ws);             // 8 MiB
  unsigned short* WqT    = (unsigned short*)(ws +  8 * MiB);  // 2 MiB
  unsigned short* WkT    = (unsigned short*)(ws + 10 * MiB);  // 2 MiB
  unsigned short* WvT    = (unsigned short*)(ws + 12 * MiB);  // 2 MiB
  unsigned short* WoT    = (unsigned short*)(ws + 14 * MiB);  // 2 MiB
  unsigned short* q_bf   = (unsigned short*)(ws + 16 * MiB);  // 8 MiB
  unsigned short* k_all  = (unsigned short*)(ws + 24 * MiB);  // 12 MiB
  unsigned short* v_all  = (unsigned short*)(ws + 36 * MiB);  // 12 MiB
  unsigned short* ctx_bf = (unsigned short*)(ws + 48 * MiB);  // 8 MiB

  convert_hidden_kernel<<<dim3((B_ * L_ * D_) / 256), 256, 0, stream>>>(hidden, hid_bf);
  convert_wt_kernel<<<dim3((4 * D_ * HD_) / 256), 256, 0, stream>>>(
      Wq, Wk, Wv, Wo, WqT, WkT, WvT, WoT);
  cache_fill_kernel<<<dim3((2 * B_ * H_ * (SEC_ + XL_) * DK_) / 256), 256, 0, stream>>>(
      xl_k, xl_v, sec_k, sec_v, k_all, v_all);

  gemm_qkv_kernel<<<dim3(HD_ / 256, (B_ * L_) / 128, 3), 256, 0, stream>>>(
      hid_bf, WqT, WkT, WvT, q_bf, k_all, v_all);

  attn_kernel<<<dim3(L_ / 128, H_, B_), 256, 0, stream>>>(
      q_bf, k_all, v_all, doc_ids, xl_doc, sec_doc, relb, secbias, ctx_bf);

  gemm_out_kernel<<<dim3(D_ / 256, (B_ * L_) / 128, 1), 256, 0, stream>>>(
      ctx_bf, WoT, out);
}
